// MultiHeadAttention_24034636988722
// MI455X (gfx1250) — matrix operand order repaired and checked by reading, not compiled, not run on hardware
//
#include <hip/hip_runtime.h>
#include <hip/hip_bf16.h>

// ---------------------------------------------------------------------------
// GQA attention block for MI455X (gfx1250, wave32, WMMA bf16 path)
//   qkv = x @ Wqkv ; RoPE(q,k) ; causal flash attention ; out = attn @ Wproj + b
// Matmuls: v_wmma_f32_16x16x32_bf16 (fp32 accum).
// Data movement: double-buffered GLOBAL_LOAD_ASYNC_TO_LDS_B128 (ASYNCcnt).
// ---------------------------------------------------------------------------

typedef __bf16 bf16_t;
typedef __attribute__((ext_vector_type(16))) __bf16 v16bf;
typedef __attribute__((ext_vector_type(8)))  __bf16 v8bf;
typedef __attribute__((ext_vector_type(8)))  float  v8f;

#define BATCH     4
#define SEQ       2048
#define EMBD      1024
#define N_HEAD    16
#define KV_HEADS  4
#define Q_PER_KV  4
#define HEAD_DIM  64
#define QKV_N     (EMBD + 2 * KV_HEADS * HEAD_DIM)   // 1536
#define ROWS      (BATCH * SEQ)                       // 8192

// GEMM tiling: block 128x128, K-step 64, double-buffered LDS
#define BM 128
#define BN 128
#define BK 64
#define KSTRIDE 72   // 64 data + 8 pad bf16 (144B row stride, 16B aligned)

// Attention tiling
#define QT 128       // query rows per block (8 waves x 16)
#define KT 64        // kv rows per tile
#define KPAD 72

__device__ __forceinline__ v16bf make16(v8bf lo, v8bf hi) {
    union { v16bf v; v8bf h[2]; } u;
    u.h[0] = lo; u.h[1] = hi;
    return u.v;
}

__device__ __forceinline__ v8f wmma_bf16(v16bf a, v16bf b, v8f c) {
    return __builtin_amdgcn_wmma_f32_16x16x32_bf16(
        /*neg_a=*/false, a, /*neg_b=*/false, b,
        /*c_mod=*/(short)0, c, /*reuse_a=*/false, /*reuse_b=*/false);
}

// Async 16B copy global -> LDS (per lane). VDST = 32-bit LDS byte address,
// address = 64-bit VGPR pair. Tracked by ASYNCcnt, not LOADcnt/DScnt.
__device__ __forceinline__ void async_ld16(const void* gp, const void* lp) {
    uint64_t ga = (uint64_t)(uintptr_t)gp;
    uint32_t la = (uint32_t)(uintptr_t)(__attribute__((address_space(3))) const char*)lp;
    asm volatile("global_load_async_to_lds_b128 %0, %1, off"
                 :: "v"(la), "v"(ga)
                 : "memory");
}

#define WAIT_ASYNC(n) asm volatile("s_wait_asynccnt " #n ::: "memory")

// ---------------------------------------------------------------------------
// fp32 -> bf16 elementwise conversion
// ---------------------------------------------------------------------------
__global__ void cvt_bf16_kernel(const float* __restrict__ in,
                                bf16_t* __restrict__ out, int n) {
    int i = blockIdx.x * blockDim.x + threadIdx.x;
    if (i < n) out[i] = (bf16_t)in[i];
}

// fp32 [K][N] -> bf16 [N][K] (weights, so B fragments read contiguous K)
__global__ void transpose_cvt_kernel(const float* __restrict__ in,
                                     bf16_t* __restrict__ out, int K, int N) {
    int i = blockIdx.x * blockDim.x + threadIdx.x;
    if (i < K * N) {
        int k = i / N, n = i % N;
        out[(size_t)n * K + k] = (bf16_t)in[i];
    }
}

// ---------------------------------------------------------------------------
// bf16 GEMM: C[M][N] = A[M][K] * Bt[N][K]^T (+ bias), fp32 output
// 256 threads = 8 waves; block tile 128x128x64; wave tile 64x32.
// Double-buffered: async-copy tile i+1 while WMMAs consume tile i.
// ---------------------------------------------------------------------------
__global__ __launch_bounds__(256) void gemm_bf16_kernel(
    const bf16_t* __restrict__ A,
    const bf16_t* __restrict__ Bt,
    float* __restrict__ C,
    const float* __restrict__ bias,
    int M, int N, int K) {
    __shared__ bf16_t As[2][BM][KSTRIDE];
    __shared__ bf16_t Bs[2][BN][KSTRIDE];

    const int tid  = threadIdx.x;
    const int lane = tid & 31;
    const int wave = tid >> 5;
    const int wrow = wave >> 2;      // 0..1 : 64-row slab
    const int wcol = wave & 3;       // 0..3 : 32-col slab
    const int bm   = blockIdx.y * BM;
    const int bn   = blockIdx.x * BN;
    const int lm   = lane & 15;
    const int lhi  = lane >> 4;

    const v8f zero8 = {0.f, 0.f, 0.f, 0.f, 0.f, 0.f, 0.f, 0.f};
    v8f acc[4][2];
#pragma unroll
    for (int i = 0; i < 4; ++i)
#pragma unroll
        for (int j = 0; j < 2; ++j) acc[i][j] = zero8;

    const int nstep = K / BK;

    // stage tile 0 into buffer 0: 1024 16B-chunks per matrix, 4 per thread,
    // 8 async instructions per wave per tile.
#pragma unroll
    for (int it = 0; it < 4; ++it) {
        int chunk = it * 256 + tid;          // r = chunk/8, 8 chunks per row
        int r = chunk >> 3, cc = chunk & 7;
        async_ld16(&A[(size_t)(bm + r) * K + cc * 8], &As[0][r][cc * 8]);
        async_ld16(&Bt[(size_t)(bn + r) * K + cc * 8], &Bs[0][r][cc * 8]);
    }

    for (int step = 0; step < nstep; ++step) {
        const int buf = step & 1;
        if (step + 1 < nstep) {
            const int k1 = (step + 1) * BK;
#pragma unroll
            for (int it = 0; it < 4; ++it) {
                int chunk = it * 256 + tid;
                int r = chunk >> 3, cc = chunk & 7;
                async_ld16(&A[(size_t)(bm + r) * K + k1 + cc * 8],
                           &As[buf ^ 1][r][cc * 8]);
                async_ld16(&Bt[(size_t)(bn + r) * K + k1 + cc * 8],
                           &Bs[buf ^ 1][r][cc * 8]);
            }
            // current tile's 8 wave-ops complete once <=8 remain outstanding
            WAIT_ASYNC(8);
        } else {
            WAIT_ASYNC(0);
        }
        if (step + 2 < nstep) {   // nudge tile i+2 toward L2
            __builtin_prefetch(&A[(size_t)(bm + (tid >> 1)) * K + (step + 2) * BK], 0, 1);
            __builtin_prefetch(&Bt[(size_t)(bn + (tid >> 1)) * K + (step + 2) * BK], 0, 1);
        }
        __syncthreads();

#pragma unroll
        for (int kh = 0; kh < 2; ++kh) {
            // A frags: lanes0-15 K{0..7,16..23}, lanes16-31 K{8..15,24..31}
            v16bf af[4], bfg[2];
#pragma unroll
            for (int i = 0; i < 4; ++i) {
                const bf16_t* p =
                    &As[buf][wrow * 64 + i * 16 + lm][kh * 32 + lhi * 8];
                af[i] = make16(*(const v8bf*)p, *(const v8bf*)(p + 16));
            }
            // B frags: lanes0-15 K{0..7,16..23}, lanes16-31 K{8..15,24..31}
#pragma unroll
            for (int j = 0; j < 2; ++j) {
                const bf16_t* p =
                    &Bs[buf][wcol * 32 + j * 16 + lm][kh * 32 + lhi * 8];
                bfg[j] = make16(*(const v8bf*)p, *(const v8bf*)(p + 16));
            }
#pragma unroll
            for (int i = 0; i < 4; ++i)
#pragma unroll
                for (int j = 0; j < 2; ++j)
                    acc[i][j] = wmma_bf16(af[i], bfg[j], acc[i][j]);
        }
        __syncthreads();   // protect buf before it is refilled at step+2
    }

    // epilogue: C layout row = jv + 8*lhi, col = lm (within 16x16 tile)
#pragma unroll
    for (int i = 0; i < 4; ++i)
#pragma unroll
        for (int j = 0; j < 2; ++j) {
            int col = bn + wcol * 32 + j * 16 + lm;
            float bv = bias ? bias[col] : 0.0f;
#pragma unroll
            for (int jv = 0; jv < 8; ++jv) {
                int row = bm + wrow * 64 + i * 16 + lhi * 8 + jv;
                C[(size_t)row * N + col] = acc[i][j][jv] + bv;
            }
        }
}

// ---------------------------------------------------------------------------
// RoPE + scatter: qkv fp32 [ROWS][1536] ->
//   qb [B][H][T][64] bf16 (rotated, * 1/sqrt(64))
//   kb [B][KVH][T][64] bf16 (rotated), vb [B][KVH][T][64] bf16
// ---------------------------------------------------------------------------
__global__ void rope_scatter_kernel(const float* __restrict__ qkv,
                                    bf16_t* __restrict__ qb,
                                    bf16_t* __restrict__ kb,
                                    bf16_t* __restrict__ vb) {
    const int row = blockIdx.x;              // 0..ROWS-1
    const int b = row / SEQ, t = row % SEQ;
    const float* src = qkv + (size_t)row * QKV_N;

    for (int p = threadIdx.x; p < (N_HEAD + KV_HEADS) * 32; p += blockDim.x) {
        int head = p >> 5;
        int i = p & 31;
        float ang = (float)t * __powf(10000.0f, -(float)(2 * i) / 64.0f);
        float c, s;
        __sincosf(ang, &s, &c);
        if (head < N_HEAD) {
            float x1 = src[head * 64 + 2 * i];
            float x2 = src[head * 64 + 2 * i + 1];
            bf16_t* dst = qb + (((size_t)b * N_HEAD + head) * SEQ + t) * 64;
            dst[i]      = (bf16_t)(0.125f * (x1 * c - x2 * s));   // 1/sqrt(64)
            dst[i + 32] = (bf16_t)(0.125f * (x1 * s + x2 * c));
        } else {
            int kh = head - N_HEAD;
            const float* ks = src + EMBD + kh * 64;
            float x1 = ks[2 * i], x2 = ks[2 * i + 1];
            bf16_t* dst = kb + (((size_t)b * KV_HEADS + kh) * SEQ + t) * 64;
            dst[i]      = (bf16_t)(x1 * c - x2 * s);
            dst[i + 32] = (bf16_t)(x1 * s + x2 * c);
        }
    }
    for (int e = threadIdx.x; e < KV_HEADS * HEAD_DIM; e += blockDim.x) {
        int kh = e >> 6, d = e & 63;
        vb[(((size_t)b * KV_HEADS + kh) * SEQ + t) * 64 + d] =
            (bf16_t)src[EMBD + KV_HEADS * HEAD_DIM + e];
    }
}

// ---------------------------------------------------------------------------
// Causal flash attention, one block per (qtile, head, batch).
// K tile staged via async-to-LDS; V staged through VGPRs (transposed write).
// Online softmax in fp32 with 16-lane shuffles; P -> LDS -> A-frag relayout.
// ---------------------------------------------------------------------------
__global__ __launch_bounds__(256) void attn_kernel(
    const bf16_t* __restrict__ qb, const bf16_t* __restrict__ kb,
    const bf16_t* __restrict__ vb, bf16_t* __restrict__ ob) {
    __shared__ bf16_t Ks[KT][KPAD];           // [kv][dim]
    __shared__ bf16_t Vts[HEAD_DIM][KPAD];    // [dim][kv]  (V transposed)
    __shared__ bf16_t Ps[8][16][HEAD_DIM];    // wave-private P scratch

    const int tid  = threadIdx.x;
    const int lane = tid & 31;
    const int w    = tid >> 5;
    const int lm   = lane & 15;
    const int lhi  = lane >> 4;
    const int qtile = blockIdx.x, h = blockIdx.y, b = blockIdx.z;
    const int kvh  = h / Q_PER_KV;
    const int qbase = qtile * QT + w * 16;    // sequence index of this wave

    const bf16_t* qpl = qb + ((size_t)b * N_HEAD + h) * SEQ * 64;
    const bf16_t* kpl = kb + ((size_t)b * KV_HEADS + kvh) * SEQ * 64;
    const bf16_t* vpl = vb + ((size_t)b * KV_HEADS + kvh) * SEQ * 64;

    // Q fragments (A layout), held for the whole block
    v16bf qa[2];
#pragma unroll
    for (int kh = 0; kh < 2; ++kh) {
        const bf16_t* p = qpl + (size_t)(qbase + lm) * 64 + kh * 32 + lhi * 8;
        qa[kh] = make16(*(const v8bf*)p, *(const v8bf*)(p + 16));
    }

    const v8f zero8 = {0.f, 0.f, 0.f, 0.f, 0.f, 0.f, 0.f, 0.f};
    v8f o[4];
    float mrow[8], lrow[8];
#pragma unroll
    for (int c = 0; c < 4; ++c) o[c] = zero8;
#pragma unroll
    for (int jv = 0; jv < 8; ++jv) { mrow[jv] = -1e30f; lrow[jv] = 0.f; }

    const int nkt = (qtile + 1) * (QT / KT);  // causal extent in 64-row tiles
    for (int kt = 0; kt < nkt; ++kt) {
        const int kt0 = kt * KT;
        // K tile: async global->LDS (512 chunks of 16B, 2 per thread)
        // V tile: through VGPRs with transposed LDS scatter
#pragma unroll
        for (int it = 0; it < 2; ++it) {
            int chunk = it * 256 + tid;
            int r = chunk >> 3, cc = chunk & 7;
            async_ld16(&kpl[(size_t)(kt0 + r) * 64 + cc * 8], &Ks[r][cc * 8]);
            uint4 vv = *(const uint4*)&vpl[(size_t)(kt0 + r) * 64 + cc * 8];
            const bf16_t* ve = (const bf16_t*)&vv;
#pragma unroll
            for (int e = 0; e < 8; ++e) Vts[cc * 8 + e][r] = ve[e];
        }
        WAIT_ASYNC(0);
        __syncthreads();

        // S = Q @ K^T : 16x64 per wave, inner dim 64 (2 WMMAs per 16x16 tile)
        v8f s[4];
#pragma unroll
        for (int tc = 0; tc < 4; ++tc) {
            s[tc] = zero8;
#pragma unroll
            for (int kh = 0; kh < 2; ++kh) {
                const bf16_t* p = &Ks[tc * 16 + lm][kh * 32 + lhi * 8];
                v16bf kf = make16(*(const v8bf*)p, *(const v8bf*)(p + 16));
                s[tc] = wmma_bf16(qa[kh], kf, s[tc]);
            }
        }

        // causal mask (skip if tile is entirely below the diagonal)
        if (kt0 + KT - 1 > qbase) {
#pragma unroll
            for (int tc = 0; tc < 4; ++tc) {
                int col = kt0 + tc * 16 + lm;
#pragma unroll
                for (int jv = 0; jv < 8; ++jv) {
                    int rowg = qbase + lhi * 8 + jv;
                    if (col > rowg) s[tc][jv] = -1e30f;
                }
            }
        }

        // online softmax per row (row = jv + 8*lhi; reduce across 16 lanes)
        float alpha[8];
#pragma unroll
        for (int jv = 0; jv < 8; ++jv) {
            float sm = fmaxf(fmaxf(s[0][jv], s[1][jv]),
                             fmaxf(s[2][jv], s[3][jv]));
#pragma unroll
            for (int off = 1; off < 16; off <<= 1)
                sm = fmaxf(sm, __shfl_xor(sm, off));
            float mnew = fmaxf(mrow[jv], sm);
            alpha[jv] = __expf(mrow[jv] - mnew);
            mrow[jv] = mnew;
            float ps = 0.f;
#pragma unroll
            for (int tc = 0; tc < 4; ++tc) {
                float pv = __expf(s[tc][jv] - mnew);
                s[tc][jv] = pv;
                ps += pv;
            }
#pragma unroll
            for (int off = 1; off < 16; off <<= 1)
                ps += __shfl_xor(ps, off);
            lrow[jv] = lrow[jv] * alpha[jv] + ps;
#pragma unroll
            for (int c = 0; c < 4; ++c) o[c][jv] *= alpha[jv];
        }

        // C-layout -> A-layout for P via wave-private LDS (in-order DS ops)
#pragma unroll
        for (int tc = 0; tc < 4; ++tc)
#pragma unroll
            for (int jv = 0; jv < 8; ++jv)
                Ps[w][lhi * 8 + jv][tc * 16 + lm] = (bf16_t)s[tc][jv];

        v16bf pa[2];
#pragma unroll
        for (int kh = 0; kh < 2; ++kh) {
            const bf16_t* p = &Ps[w][lm][kh * 32 + lhi * 8];
            pa[kh] = make16(*(const v8bf*)p, *(const v8bf*)(p + 16));
        }

        // O += P @ V : B fragment = columns of V = rows of V^T
#pragma unroll
        for (int c = 0; c < 4; ++c)
#pragma unroll
            for (int kh = 0; kh < 2; ++kh) {
                const bf16_t* p = &Vts[c * 16 + lm][kh * 32 + lhi * 8];
                v16bf vf = make16(*(const v8bf*)p, *(const v8bf*)(p + 16));
                o[c] = wmma_bf16(pa[kh], vf, o[c]);
            }
        __syncthreads();
    }

    // normalize and scatter to [b*SEQ + t][h*64 + d] bf16
    float inv[8];
#pragma unroll
    for (int jv = 0; jv < 8; ++jv) inv[jv] = 1.0f / lrow[jv];
#pragma unroll
    for (int c = 0; c < 4; ++c)
#pragma unroll
        for (int jv = 0; jv < 8; ++jv) {
            int tpos = qbase + lhi * 8 + jv;
            ob[((size_t)b * SEQ + tpos) * EMBD + h * HEAD_DIM + c * 16 + lm] =
                (bf16_t)(o[c][jv] * inv[jv]);
        }
}

// ---------------------------------------------------------------------------
// Host-side launch
// ---------------------------------------------------------------------------
extern "C" void kernel_launch(void* const* d_in, const int* in_sizes, int n_in,
                              void* d_out, int out_size, void* d_ws,
                              size_t ws_size, hipStream_t stream) {
    const float* x      = (const float*)d_in[0];
    const float* w_qkv  = (const float*)d_in[1];
    const float* w_proj = (const float*)d_in[2];
    const float* b_proj = (const float*)d_in[3];
    float* out = (float*)d_out;

    char* ws = (char*)d_ws;
    size_t off = 0;
    auto wsalloc = [&](size_t bytes) -> void* {
        void* p = (void*)(ws + off);
        off += (bytes + 255) & ~(size_t)255;
        return p;
    };

    bf16_t* xb     = (bf16_t*)wsalloc((size_t)ROWS * EMBD * 2);
    bf16_t* wqkvT  = (bf16_t*)wsalloc((size_t)QKV_N * EMBD * 2);
    bf16_t* wprojT = (bf16_t*)wsalloc((size_t)EMBD * EMBD * 2);
    float*  qkv    = (float*) wsalloc((size_t)ROWS * QKV_N * 4);
    bf16_t* qb_    = (bf16_t*)wsalloc((size_t)ROWS * EMBD * 2);
    bf16_t* kb_    = (bf16_t*)wsalloc((size_t)BATCH * KV_HEADS * SEQ * 64 * 2);
    bf16_t* vb_    = (bf16_t*)wsalloc((size_t)BATCH * KV_HEADS * SEQ * 64 * 2);
    bf16_t* attn   = (bf16_t*)wsalloc((size_t)ROWS * EMBD * 2);

    cvt_bf16_kernel<<<(ROWS * EMBD) / 256, 256, 0, stream>>>(x, xb, ROWS * EMBD);
    transpose_cvt_kernel<<<(EMBD * QKV_N) / 256, 256, 0, stream>>>(
        w_qkv, wqkvT, EMBD, QKV_N);
    transpose_cvt_kernel<<<(EMBD * EMBD) / 256, 256, 0, stream>>>(
        w_proj, wprojT, EMBD, EMBD);

    gemm_bf16_kernel<<<dim3(QKV_N / BN, ROWS / BM), 256, 0, stream>>>(
        xb, wqkvT, qkv, nullptr, ROWS, QKV_N, EMBD);

    rope_scatter_kernel<<<ROWS, 256, 0, stream>>>(qkv, qb_, kb_, vb_);

    attn_kernel<<<dim3(SEQ / QT, N_HEAD, BATCH), 256, 0, stream>>>(
        qb_, kb_, vb_, attn);

    gemm_bf16_kernel<<<dim3(EMBD / BN, ROWS / BM), 256, 0, stream>>>(
        attn, wprojT, out, b_proj, ROWS, EMBD, EMBD);
}
